// Mini_pointgnn_v1_67310727463235
// MI455X (gfx1250) — matrix operand order repaired and checked by reading, not compiled, not run on hardware
//
#include <hip/hip_runtime.h>
#include <hip/hip_bf16.h>
#include <math.h>

// ---------------- constants from the reference ----------------
// NOTE: NPTS, MCLS, E0N, E1N are all exact multiples of 16 -> no partial
// 16-row tiles anywhere; all tile guards are statically removed.
#define NPTS 100000
#define MCLS 10000
#define E0N  1600000
#define E1N  320000
#define DDIM 64
#define CIN  4
#define NCLS 4

typedef __attribute__((ext_vector_type(16))) _Float16 v16h;
typedef __attribute__((ext_vector_type(8)))  _Float16 v8h;
typedef __attribute__((ext_vector_type(8)))  float    v8f;

union H16 {
    v16h v;
    v8h  h8[2];
};

// ---------------- weight fp32 -> f16, fragment-major packed ----------------
// dst[(((nt*NK + kt)*2 + hi)*16 + col)*16 + h] = W[(kt*32 + hi*16 + h)*64 + nt*16 + col]
// (zero-padded for k >= kRows). Each lane's B fragment = 32 contiguous bytes.
__global__ void pack_b_kernel(const float* __restrict__ src, _Float16* __restrict__ dst,
                              int kRows, int NK) {
    int total = 4 * NK * 512;
    for (int i = blockIdx.x * blockDim.x + threadIdx.x; i < total; i += gridDim.x * blockDim.x) {
        int h   = i & 15;
        int col = (i >> 4) & 15;
        int hi  = (i >> 8) & 1;
        int rest = i >> 9;
        int kt = rest % NK, nt = rest / NK;
        int k = kt * 32 + hi * 16 + h;
        int n = nt * 16 + col;
        dst[i] = (k < kRows) ? (_Float16)src[k * DDIM + n] : (_Float16)0.0f;
    }
}

__global__ void fill0_kernel(int* __restrict__ p, long n) {
    for (long i = blockIdx.x * (long)blockDim.x + threadIdx.x; i < n; i += (long)gridDim.x * blockDim.x)
        p[i] = 0;
}

__global__ void add_kernel(float* __restrict__ a, const float* __restrict__ b, long n) {
    for (long i = blockIdx.x * (long)blockDim.x + threadIdx.x; i < n; i += (long)gridDim.x * blockDim.x)
        a[i] += b[i];
}

// rel = points - centers[labels]
__global__ void rel_kernel(const float* __restrict__ pts, const float* __restrict__ ctr,
                           const int* __restrict__ labels, float* __restrict__ rel, int n) {
    int i = blockIdx.x * blockDim.x + threadIdx.x;
    if (i >= n) return;
    int l = labels[i];
    rel[3 * i + 0] = pts[3 * i + 0] - ctr[3 * l + 0];
    rel[3 * i + 1] = pts[3 * i + 1] - ctr[3 * l + 1];
    rel[3 * i + 2] = pts[3 * i + 2] - ctr[3 * l + 2];
}

// delta = tanh(x @ Wh + bh), 64 -> 3 (tiny, scalar)
__global__ void delta_kernel(const float* __restrict__ X, const float* __restrict__ Wh,
                             const float* __restrict__ bh, float* __restrict__ out, int n) {
    int i = blockIdx.x * blockDim.x + threadIdx.x;
    if (i >= n) return;
    float s0 = bh[0], s1 = bh[1], s2 = bh[2];
    const float* xr = X + (long)i * DDIM;
#pragma unroll 8
    for (int d = 0; d < DDIM; ++d) {
        float x = xr[d];
        s0 += x * Wh[d * 3 + 0];
        s1 += x * Wh[d * 3 + 1];
        s2 += x * Wh[d * 3 + 2];
    }
    out[3 * i + 0] = tanhf(s0);
    out[3 * i + 1] = tanhf(s1);
    out[3 * i + 2] = tanhf(s2);
}

// segment-max scatter (values >= 0, 0-initialized target; int-max == float-max)
__global__ void seg_scatter_kernel(const float* __restrict__ h, const int* __restrict__ labels,
                                   int* __restrict__ outBits, long total) {
    for (long i = blockIdx.x * (long)blockDim.x + threadIdx.x; i < total; i += (long)gridDim.x * blockDim.x) {
        int n = (int)(i >> 6), c = (int)(i & 63);
        float v = fmaxf(h[i], 0.0f);
        atomicMax(&outBits[(long)labels[n] * DDIM + c], __float_as_int(v));
    }
}

// final classifier: out = t @ W2 + b2  (64 -> 4, tiny)
__global__ void cls2_kernel(const float* __restrict__ t, const float* __restrict__ W2,
                            const float* __restrict__ b2, float* __restrict__ out, int n) {
    int i = blockIdx.x * blockDim.x + threadIdx.x;
    if (i >= n) return;
    float s[NCLS];
#pragma unroll
    for (int j = 0; j < NCLS; ++j) s[j] = b2[j];
    const float* tr = t + (long)i * DDIM;
#pragma unroll 8
    for (int d = 0; d < DDIM; ++d) {
        float x = tr[d];
#pragma unroll
        for (int j = 0; j < NCLS; ++j) s[j] += x * W2[d * NCLS + j];
    }
#pragma unroll
    for (int j = 0; j < NCLS; ++j) out[i * NCLS + j] = s[j];
}

// ---------------- generic WMMA node GEMM (B fragments register-resident) ----
// out[row,:64] = act( concat(X[gather(row)], X2[row]) @ W + bias ) (+ skip)
// nRows must be a multiple of 16 (true for all call sites).
template <int KPAD>
__global__ __launch_bounds__(256) void node_gemm_kernel(
    const float* __restrict__ X, int xCols,
    const int* __restrict__ rowIdx,
    const float* __restrict__ X2, int x2Cols,
    const _Float16* __restrict__ Wp,  // fragment-major packed (pack_b_kernel)
    const float* __restrict__ bias,
    const float* __restrict__ skip, int doRelu,
    float* __restrict__ out, int nRows) {
    constexpr int NK = KPAD / 32;
    __shared__ __align__(16) _Float16 ldsA[8][16 * KPAD];
    const int tid = threadIdx.x, lane = tid & 31, wave = tid >> 5;
    const int col = lane & 15, hi = lane >> 4;

    // load all B fragments into registers once (32B contiguous per fragment)
    H16 bfr[4][NK];
#pragma unroll
    for (int nt = 0; nt < 4; ++nt)
#pragma unroll
        for (int kt = 0; kt < NK; ++kt) {
            const v8h* p = (const v8h*)(Wp + (((nt * NK + kt) * 2 + hi) * 16 + col) * 16);
            bfr[nt][kt].h8[0] = p[0];
            bfr[nt][kt].h8[1] = p[1];
        }
    // hoist bias loads out of the tile loop
    float bnt[4];
#pragma unroll
    for (int nt = 0; nt < 4; ++nt) bnt[nt] = bias ? bias[nt * 16 + col] : 0.0f;

    const int nTiles = nRows >> 4;
    for (int tile = blockIdx.x * 8 + wave; tile < nTiles; tile += gridDim.x * 8) {
        const int base = tile << 4;
        // stage 16xKPAD A tile into LDS as f16 (coalesced on k)
        for (int r = 0; r < 16; ++r) {
            int row = base + r;
            int src = rowIdx ? rowIdx[row] : row;
#pragma unroll
            for (int k = lane; k < KPAD; k += 32) {
                float v = 0.0f;
                if (k < xCols) v = X[(long)src * xCols + k];
                else if (X2 && k < xCols + x2Cols) v = X2[(long)row * x2Cols + (k - xCols)];
                ldsA[wave][r * KPAD + k] = (_Float16)v;
            }
        }
        asm volatile("s_wait_dscnt 0" ::: "memory");  // same-wave LDS RAW

        v8f acc[4];
#pragma unroll
        for (int nt = 0; nt < 4; ++nt)
#pragma unroll
            for (int v = 0; v < 8; ++v) acc[nt][v] = bnt[nt];
        const int arow = lane & 15;
#pragma unroll
        for (int kt = 0; kt < NK; ++kt) {
            // A fragment: two contiguous 16B runs per lane -> 2x ds_load_b128
            const _Float16* ap = &ldsA[wave][arow * KPAD + kt * 32 + hi * 16];
            H16 a;
            a.h8[0] = *(const v8h*)ap;
            a.h8[1] = *(const v8h*)(ap + 8);
#pragma unroll
            for (int nt = 0; nt < 4; ++nt)
                acc[nt] = __builtin_amdgcn_wmma_f32_16x16x32_f16(
                    false, a.v, false, bfr[nt][kt].v, (short)0, acc[nt], false, false);
        }
#pragma unroll
        for (int nt = 0; nt < 4; ++nt) {
#pragma unroll
            for (int v = 0; v < 8; ++v) {
                int row = base + v + hi * 8;
                float val = acc[nt][v];
                if (doRelu) val = fmaxf(val, 0.0f);
                if (skip) val += skip[(long)row * DDIM + nt * 16 + col];
                out[(long)row * DDIM + nt * 16 + col] = val;
            }
        }
    }
}

// ---------------- edge MLP + atomic segment-max ----------------
// per 16-edge tile: e = relu([pos[s]-pos[d]+delta[d], x[s]] @ Wf + bf); agg[d] = max
// nE must be a multiple of 16 (true for E0N and E1N).
__global__ __launch_bounds__(256) void edge_mlp_kernel(
    const float* __restrict__ X, const float* __restrict__ pos,
    const float* __restrict__ delta, const int* __restrict__ edges,
    const _Float16* __restrict__ Wp /*packed 96x64*/, const float* __restrict__ bias,
    int* __restrict__ aggBits, long nE) {
    constexpr int KPAD = 96, NK = 3;
    __shared__ __align__(16) _Float16 ldsA[8][16 * KPAD];
    __shared__ int ldsD[8][16];  // destination node per edge row
    const int tid = threadIdx.x, lane = tid & 31, wave = tid >> 5;
    const int col = lane & 15, hi = lane >> 4;

    H16 bfr[4][NK];
#pragma unroll
    for (int nt = 0; nt < 4; ++nt)
#pragma unroll
        for (int kt = 0; kt < NK; ++kt) {
            const v8h* p = (const v8h*)(Wp + (((nt * NK + kt) * 2 + hi) * 16 + col) * 16);
            bfr[nt][kt].h8[0] = p[0];
            bfr[nt][kt].h8[1] = p[1];
        }
    float bnt[4];
#pragma unroll
    for (int nt = 0; nt < 4; ++nt) bnt[nt] = bias[nt * 16 + col];

    const long nTiles = nE >> 4;
    const long stride = (long)gridDim.x * 8;
    for (long tile = (long)blockIdx.x * 8 + wave; tile < nTiles; tile += stride) {
        const long base = tile << 4;
        if (tile + stride < nTiles)
            __builtin_prefetch(&edges[2 * ((tile + stride) << 4)], 0, 0);
        for (int r = 0; r < 16; ++r) {
            long e = base + r;
            int s = edges[2 * e], d = edges[2 * e + 1];
            ldsD[wave][r] = d;  // all lanes write same value (benign)
#pragma unroll
            for (int k = lane; k < KPAD; k += 32) {
                float v = 0.0f;
                if (k < 3) v = pos[3 * s + k] - pos[3 * d + k] + delta[3 * d + k];
                else if (k < 3 + DDIM) v = X[(long)s * DDIM + (k - 3)];
                ldsA[wave][r * KPAD + k] = (_Float16)v;
            }
        }
        asm volatile("s_wait_dscnt 0" ::: "memory");

        v8f acc[4];
#pragma unroll
        for (int nt = 0; nt < 4; ++nt)
#pragma unroll
            for (int v = 0; v < 8; ++v) acc[nt][v] = bnt[nt];
        const int arow = lane & 15;
#pragma unroll
        for (int kt = 0; kt < NK; ++kt) {
            const _Float16* ap = &ldsA[wave][arow * KPAD + kt * 32 + hi * 16];
            H16 a;
            a.h8[0] = *(const v8h*)ap;
            a.h8[1] = *(const v8h*)(ap + 8);
#pragma unroll
            for (int nt = 0; nt < 4; ++nt)
                acc[nt] = __builtin_amdgcn_wmma_f32_16x16x32_f16(
                    false, a.v, false, bfr[nt][kt].v, (short)0, acc[nt], false, false);
        }
        // destinations from LDS; unconditional atomic scatter (no partial tiles)
        int dsts[8];
#pragma unroll
        for (int v = 0; v < 8; ++v) dsts[v] = ldsD[wave][v + hi * 8];
#pragma unroll
        for (int nt = 0; nt < 4; ++nt) {
#pragma unroll
            for (int v = 0; v < 8; ++v) {
                float val = fmaxf(acc[nt][v], 0.0f);
                atomicMax(&aggBits[(long)dsts[v] * DDIM + nt * 16 + col],
                          __float_as_int(val));
            }
        }
    }
}

// ---------------- host orchestration ----------------
static inline size_t alignUp(size_t x) { return (x + 255) & ~(size_t)255; }

extern "C" void kernel_launch(void* const* d_in, const int* in_sizes, int n_in,
                              void* d_out, int out_size, void* d_ws, size_t ws_size,
                              hipStream_t stream) {
    const float* features = (const float*)d_in[0];
    const float* points   = (const float*)d_in[1];
    const float* centers  = (const float*)d_in[2];
    const int*   labels   = (const int*)d_in[3];
    const int*   l0e      = (const int*)d_in[4];
    const int*   l1e      = (const int*)d_in[5];
    const float* fe_W  = (const float*)d_in[6];
    const float* fe_b  = (const float*)d_in[7];
    const float* Wh    = (const float*)d_in[8];
    const float* bh    = (const float*)d_in[9];
    const float* Wf    = (const float*)d_in[10];
    const float* bf    = (const float*)d_in[11];
    const float* Wg    = (const float*)d_in[12];
    const float* bg    = (const float*)d_in[13];
    const float* m2l_W = (const float*)d_in[14];
    const float* m2l_b = (const float*)d_in[15];
    const float* l2m_W = (const float*)d_in[16];
    const float* l2m_b = (const float*)d_in[17];
    const float* cW1   = (const float*)d_in[18];
    const float* cb1   = (const float*)d_in[19];
    const float* cW2   = (const float*)d_in[20];
    const float* cb2   = (const float*)d_in[21];

    char* base = (char*)d_ws;
    size_t off = 0;
    auto alloc = [&](size_t bytes) -> void* {
        void* r = base + off;
        off = alignUp(off + bytes);
        return r;
    };
    _Float16* feW16  = (_Float16*)alloc(32 * DDIM * 2);
    _Float16* Wf16_0 = (_Float16*)alloc(96 * DDIM * 2);
    _Float16* Wf16_1 = (_Float16*)alloc(96 * DDIM * 2);
    _Float16* Wf16_2 = (_Float16*)alloc(96 * DDIM * 2);
    _Float16* Wg16_0 = (_Float16*)alloc(64 * DDIM * 2);
    _Float16* Wg16_1 = (_Float16*)alloc(64 * DDIM * 2);
    _Float16* Wg16_2 = (_Float16*)alloc(64 * DDIM * 2);
    _Float16* m2l16  = (_Float16*)alloc(96 * DDIM * 2);
    _Float16* l2m16  = (_Float16*)alloc(96 * DDIM * 2);
    _Float16* cW1_16 = (_Float16*)alloc(64 * DDIM * 2);
    float* rel    = (float*)alloc((size_t)NPTS * 3 * 4);
    float* delta  = (float*)alloc((size_t)NPTS * 3 * 4);
    float* bufA   = (float*)alloc((size_t)NPTS * DDIM * 4);
    float* bufB   = (float*)alloc((size_t)NPTS * DDIM * 4);
    float* bufC   = (float*)alloc((size_t)NPTS * DDIM * 4);
    float* cbuf   = (float*)alloc((size_t)MCLS * DDIM * 4);
    float* cdelta = (float*)alloc((size_t)MCLS * 3 * 4);
    float* cagg   = (float*)alloc((size_t)MCLS * DDIM * 4);
    float* c4buf  = (float*)alloc((size_t)MCLS * DDIM * 4);
    if (off > ws_size) return;  // workspace too small

    const int TB = 256;
    const int gN   = (NPTS + TB - 1) / TB;
    const int gM   = (MCLS + TB - 1) / TB;
    const int gGN  = ((NPTS + 15) / 16 + 7) / 8;   // node-gemm blocks (N rows)
    const int gGM  = ((MCLS + 15) / 16 + 7) / 8;
    const int gE0  = ((E0N + 15) / 16 + 7) / 8;
    const int gE1  = ((E1N + 15) / 16 + 7) / 8;
    const int gFIL = 2048, gSC = 2048;

    // --- weight prep (fragment-major f16 packing) ---
    pack_b_kernel<<<8, TB, 0, stream>>>(fe_W, feW16, CIN + 3, 1);
    pack_b_kernel<<<24, TB, 0, stream>>>(Wf + 0 * 67 * DDIM, Wf16_0, 67, 3);
    pack_b_kernel<<<24, TB, 0, stream>>>(Wf + 1 * 67 * DDIM, Wf16_1, 67, 3);
    pack_b_kernel<<<24, TB, 0, stream>>>(Wf + 2 * 67 * DDIM, Wf16_2, 67, 3);
    pack_b_kernel<<<16, TB, 0, stream>>>(Wg + 0 * 64 * DDIM, Wg16_0, 64, 2);
    pack_b_kernel<<<16, TB, 0, stream>>>(Wg + 1 * 64 * DDIM, Wg16_1, 64, 2);
    pack_b_kernel<<<16, TB, 0, stream>>>(Wg + 2 * 64 * DDIM, Wg16_2, 64, 2);
    pack_b_kernel<<<24, TB, 0, stream>>>(m2l_W, m2l16, 67, 3);
    pack_b_kernel<<<24, TB, 0, stream>>>(l2m_W, l2m16, 67, 3);
    pack_b_kernel<<<16, TB, 0, stream>>>(cW1, cW1_16, 64, 2);

    // --- rel + x1 ---
    rel_kernel<<<gN, TB, 0, stream>>>(points, centers, labels, rel, NPTS);
    node_gemm_kernel<32><<<gGN, TB, 0, stream>>>(features, CIN, nullptr, rel, 3,
                                                 feW16, fe_b, nullptr, 1, bufA, NPTS);
    // --- GNN layer 0 (points) ---
    delta_kernel<<<gN, TB, 0, stream>>>(bufA, Wh + 0, bh + 0, delta, NPTS);
    fill0_kernel<<<gFIL, TB, 0, stream>>>((int*)bufB, (long)NPTS * DDIM);
    edge_mlp_kernel<<<gE0, TB, 0, stream>>>(bufA, points, delta, l0e, Wf16_0,
                                            bf + 0, (int*)bufB, (long)E0N);
    node_gemm_kernel<64><<<gGN, TB, 0, stream>>>(bufB, DDIM, nullptr, nullptr, 0,
                                                 Wg16_0, bg + 0, bufA, 1, bufC, NPTS);  // x2
    // --- m2l + cluster pooling ---
    node_gemm_kernel<96><<<gGN, TB, 0, stream>>>(bufC, DDIM, nullptr, rel, 3,
                                                 m2l16, m2l_b, nullptr, 1, bufA, NPTS);  // h
    fill0_kernel<<<gFIL, TB, 0, stream>>>((int*)cbuf, (long)MCLS * DDIM);
    seg_scatter_kernel<<<gSC, TB, 0, stream>>>(bufA, labels, (int*)cbuf, (long)NPTS * DDIM);
    // --- GNN layer 1 (clusters) ---
    delta_kernel<<<gM, TB, 0, stream>>>(cbuf, Wh + 192, bh + 3, cdelta, MCLS);
    fill0_kernel<<<gFIL, TB, 0, stream>>>((int*)cagg, (long)MCLS * DDIM);
    edge_mlp_kernel<<<gE1, TB, 0, stream>>>(cbuf, centers, cdelta, l1e, Wf16_1,
                                            bf + 64, (int*)cagg, (long)E1N);
    node_gemm_kernel<64><<<gGM, TB, 0, stream>>>(cagg, DDIM, nullptr, nullptr, 0,
                                                 Wg16_1, bg + 64, cbuf, 1, c4buf, MCLS);  // c4
    // --- l2m (gather by labels) ---
    node_gemm_kernel<96><<<gGN, TB, 0, stream>>>(c4buf, DDIM, labels, rel, 3,
                                                 l2m16, l2m_b, nullptr, 1, bufA, NPTS);  // x5
    // --- GNN layer 2 (points) ---
    delta_kernel<<<gN, TB, 0, stream>>>(bufA, Wh + 384, bh + 6, delta, NPTS);
    fill0_kernel<<<gFIL, TB, 0, stream>>>((int*)bufB, (long)NPTS * DDIM);
    edge_mlp_kernel<<<gE0, TB, 0, stream>>>(bufA, points, delta, l0e, Wf16_2,
                                            bf + 128, (int*)bufB, (long)E0N);
    node_gemm_kernel<64><<<gGN, TB, 0, stream>>>(bufB, DDIM, nullptr, nullptr, 0,
                                                 Wg16_2, bg + 128, bufA, 1, bufB, NPTS);  // x6 (in-place ok)
    // --- final = x6 + x2; classifier ---
    add_kernel<<<gFIL, TB, 0, stream>>>(bufB, bufC, (long)NPTS * DDIM);
    node_gemm_kernel<64><<<gGN, TB, 0, stream>>>(bufB, DDIM, nullptr, nullptr, 0,
                                                 cW1_16, cb1, nullptr, 1, bufA, NPTS);  // t
    cls2_kernel<<<gN, TB, 0, stream>>>(bufA, cW2, cb2, (float*)d_out, NPTS);
}
